// DA_MoCoNNQQ_Disc_TCN_Siam_42322607735598
// MI455X (gfx1250) — matrix operand order repaired and checked by reading, not compiled, not run on hardware
//
#include <hip/hip_runtime.h>

#define DEV_INLINE __device__ __forceinline__

typedef __attribute__((ext_vector_type(16))) _Float16 v16h;
typedef __attribute__((ext_vector_type(8)))  _Float16 v8h;
typedef __attribute__((ext_vector_type(8)))  float    v8f;
typedef _Float16 half_t;

// ---------------- problem sizes ----------------
constexpr int B_   = 256;
constexpr int L_   = 1024;
constexpr int C_   = 64;
constexpr int HID_ = 256;
constexpr int KQ_  = 24576;
constexpr int WIN  = 125;     // receptive field of last timestep
constexpr int WPAD = 128;     // padded window
constexpr int RS   = 80;      // LDS row stride (halves); 32B-aligned rows
constexpr int NCOL = B_ + KQ_;   // 24832 logits columns
constexpr float INV_T = 1.0f / 0.07f;

// ---------------- d_out offsets (floats, reference return order) ----------------
constexpr size_t O_LOGITS_S  = 0;
constexpr size_t O_LBL_S     = O_LOGITS_S  + (size_t)B_ * NCOL;
constexpr size_t O_LOGITS_T  = O_LBL_S     + B_;
constexpr size_t O_LBL_T     = O_LOGITS_T  + (size_t)B_ * NCOL;
constexpr size_t O_LOGITS_TS = O_LBL_T     + B_;
constexpr size_t O_LBL_TS    = O_LOGITS_TS + (size_t)B_ * B_;
constexpr size_t O_PRED_DOM  = O_LBL_TS    + B_;
constexpr size_t O_LBL_DOM   = O_PRED_DOM  + 2 * B_;
constexpr size_t O_YS        = O_LBL_DOM   + 2 * B_;

// ---------------- WMMA helpers ----------------
DEV_INLINE v8f wmma_f16(v16h a, v16h b, v8f c) {
  // D = A(16x32 f16) * B(32x16 f16) + C(16x16 f32)
  return __builtin_amdgcn_wmma_f32_16x16x32_f16(false, a, false, b, (short)0, c, false, false);
}

// A-fragment (16x32 f16) from a row-major 64-wide f16 row.
// ISA lane layout: lane holds row M=lane%16; VGPR v holds K = 16*(v/4) + 8*half + 2*(v%4) (+1).
// Per lane this is two contiguous 8-half runs: [32ks+8h .. +7] and [32ks+16+8h .. +7].
DEV_INLINE v16h load_afrag(const half_t* __restrict__ Arow, int ks, int hh) {
  v8h lo = *(const v8h*)(Arow + 32 * ks + 8 * hh);
  v8h hi = *(const v8h*)(Arow + 32 * ks + 16 + 8 * hh);
  v16h r;
#pragma unroll
  for (int i = 0; i < 8; ++i) { r[i] = lo[i]; r[8 + i] = hi[i]; }
  return r;
}

// ---------------- kernel: pack conv weights into A-fragment order (f16) ----------------
// fragment index f = (((l*2+conv)*3 + tap)*2 + ks)*4 + mt ; 240 frags * 512 halves
__global__ __launch_bounds__(256) void k_pack_w(const float* __restrict__ c1w,
                                                const float* __restrict__ c2w,
                                                half_t* __restrict__ wfrag) {
  int t = blockIdx.x * 256 + threadIdx.x;
  if (t >= 240 * 512) return;
  int f = t >> 9;
  int e16 = t & 511;
  int lane = e16 >> 4, e = e16 & 15;
  int mt = f & 3;
  int rest = f >> 2;
  int ks = rest & 1; rest >>= 1;
  int tap = rest % 3;
  int lc = rest / 3;
  int conv = lc & 1, l = lc >> 1;
  int hh = lane >> 4, r = lane & 15;
  int v = e >> 1, p = e & 1;
  int klocal = 16 * (v >> 2) + 8 * hh + 2 * (v & 3) + p;
  int cout = mt * 16 + r;
  int cin = ks * 32 + klocal;
  const float* W = conv ? c2w : c1w;                    // (5,64,64,3)
  float val = W[((size_t)(l * 64 + cout) * 64 + cin) * 3 + tap];
  wfrag[t] = (half_t)val;
}

// ---------------- kernel: transpose queue (64 x KQ f32) -> (KQ x 64 f16) ----------------
__global__ __launch_bounds__(256) void k_pack_q(const float* __restrict__ q, half_t* __restrict__ qT) {
  __shared__ float tile[32][33];
  int jb = blockIdx.x * 32;      // along KQ
  int cb = blockIdx.y * 32;      // along C
  int tx = threadIdx.x, ty = threadIdx.y;   // block (32,8)
#pragma unroll
  for (int i = 0; i < 32; i += 8)
    tile[ty + i][tx] = q[(size_t)(cb + ty + i) * KQ_ + jb + tx];
  __syncthreads();
#pragma unroll
  for (int i = 0; i < 32; i += 8)
    qT[(size_t)(jb + ty + i) * 64 + cb + tx] = (half_t)tile[tx][ty + i];
}

// ---------------- TCN conv step (one dilated causal conv as 3 shifted WMMA GEMMs) ----------------
// D = dilation (compile time), RES = fused residual (conv2). All tile/tap guards fold at
// compile time: whole tiles that read entirely before t=0 are skipped (zero contribution);
// fully in-range tiles issue unconditional ds_load_b128; only d=1,2,4,8 nt=0 keep lane guards.
template <int D, bool RES>
DEV_INLINE void conv_step(const half_t* __restrict__ in, half_t* __restrict__ out,
                          const half_t* __restrict__ wfrag, const float* __restrict__ bias,
                          const half_t* __restrict__ resbase, int lc /* l*2+conv */,
                          int mt, int lane) {
  int hh = lane >> 4, n = lane & 15;
  v16h A[2][3];
#pragma unroll
  for (int ks = 0; ks < 2; ++ks)
#pragma unroll
    for (int tap = 0; tap < 3; ++tap) {
      int f = ((lc * 3 + tap) * 2 + ks) * 4 + mt;
      A[ks][tap] = *(const v16h*)(wfrag + (size_t)f * 512 + lane * 16);
    }
  float bvals[8];
  const float* bp = bias + mt * 16 + 8 * hh;   // bias already offset by layer
#pragma unroll
  for (int v = 0; v < 8; ++v) bvals[v] = bp[v];

#pragma unroll
  for (int nt = 0; nt < 8; ++nt) {
    int wcol = nt * 16 + n;
    v8f acc = {};
#pragma unroll
    for (int tap = 0; tap < 3; ++tap) {
      constexpr int BACKS[3] = {2 * D, D, 0};
      const int back = BACKS[tap];
      if (nt * 16 + 15 < back) continue;       // compile-time: tile entirely pre-t0 -> zero
      v16h b0, b1;
      if (nt * 16 >= back) {                   // compile-time: tile fully in range
        const half_t* rp = in + (wcol - back) * RS;
        b0 = load_afrag(rp, 0, hh);        // K slice 0..31
        b1 = load_afrag(rp, 1, hh);   // K slice 32..63
      } else {                                 // boundary tile: per-lane guard
        int wsrc = wcol - back;
        v16h z0 = {}, z1 = {};
        b0 = z0; b1 = z1;
        if (wsrc >= 0) {
          const half_t* rp = in + wsrc * RS;
          b0 = load_afrag(rp, 0, hh);
          b1 = load_afrag(rp, 1, hh);
        }
      }
      acc = wmma_f16(A[0][tap], b0, acc);
      acc = wmma_f16(A[1][tap], b1, acc);
    }
    int coff = mt * 16 + 8 * hh;
    v8h ov;
    if (RES) {
      v8h res = *(const v8h*)(resbase + wcol * RS + coff);
#pragma unroll
      for (int v = 0; v < 8; ++v) {
        float t = fmaxf(acc[v] + bvals[v], 0.f);   // relu(conv2 + b2)
        t = fmaxf(t + (float)res[v], 0.f);         // relu(h + x)
        ov[v] = (half_t)t;
      }
    } else {
#pragma unroll
      for (int v = 0; v < 8; ++v) ov[v] = (half_t)fmaxf(acc[v] + bvals[v], 0.f);
    }
    *(v8h*)(out + wcol * RS + coff) = ov;
  }
}

template <int L>
DEV_INLINE void tcn_layer(half_t* xT, half_t* hT, const half_t* wfrag,
                          const float* c1b, const float* c2b, int mt, int lane) {
  constexpr int D = 1 << L;
  conv_step<D, false>(xT, hT, wfrag, c1b + L * 64, nullptr, L * 2 + 0, mt, lane);
  __syncthreads();
  conv_step<D, true>(hT, xT, wfrag, c2b + L * 64, xT, L * 2 + 1, mt, lane);
  __syncthreads();
}

// ---------------- kernel: TCN encode last-column, l2-normalize ----------------
// blockIdx.x = s*256 + b ; s: 0=q_s 1=k_s 2=q_t 3=k_t
__global__ __launch_bounds__(128) void k_encode(
    const float* __restrict__ sq_s, const float* __restrict__ sk_s,
    const float* __restrict__ sq_t, const float* __restrict__ sk_t,
    const half_t* __restrict__ wfrag, const float* __restrict__ c1b,
    const float* __restrict__ c2b, float* __restrict__ embF32, half_t* __restrict__ embF16) {
  __shared__ alignas(32) half_t xT[WPAD * RS];
  __shared__ alignas(32) half_t hT[WPAD * RS];
  int blk = blockIdx.x;
  int s = blk >> 8, b = blk & 255;
  const float* seq = (s == 0) ? sq_s : (s == 1) ? sk_s : (s == 2) ? sq_t : sk_t;
  const float* base = seq + (size_t)b * L_ * C_ + (size_t)(L_ - WIN) * C_;
  int tid = threadIdx.x;
  for (int idx = tid; idx < WPAD * C_; idx += 128) {
    int w = idx >> 6;
    float v = (w < WIN) ? base[idx] : 0.f;   // idx = w*64 + c
    xT[w * RS + (idx & 63)] = (half_t)v;
  }
  __syncthreads();
  int lane = tid & 31;
  int mt = tid >> 5;   // one M-tile (16 output channels) per wave
  tcn_layer<0>(xT, hT, wfrag, c1b, c2b, mt, lane);
  tcn_layer<1>(xT, hT, wfrag, c1b, c2b, mt, lane);
  tcn_layer<2>(xT, hT, wfrag, c1b, c2b, mt, lane);
  tcn_layer<3>(xT, hT, wfrag, c1b, c2b, mt, lane);
  tcn_layer<4>(xT, hT, wfrag, c1b, c2b, mt, lane);
  if (mt == 0) {   // last column (w = 124) -> l2 normalize -> emb
    float v0 = (float)xT[124 * RS + lane];
    float v1 = (float)xT[124 * RS + 32 + lane];
    float ss = v0 * v0 + v1 * v1;
#pragma unroll
    for (int off = 16; off > 0; off >>= 1) ss += __shfl_xor(ss, off, 32);
    float inv = 1.f / fmaxf(sqrtf(ss), 1e-12f);
    float o0 = v0 * inv, o1 = v1 * inv;
    float* er = embF32 + (size_t)blk * 64;
    half_t* eh = embF16 + (size_t)blk * 64;
    er[lane] = o0; er[lane + 32] = o1;
    eh[lane] = (half_t)o0; eh[lane + 32] = (half_t)o1;
  }
}

// ---------------- kernel: projection MLP + l2n -> p_q (f16 rows) ----------------
__global__ __launch_bounds__(256) void k_pmlp(const float* __restrict__ emb,
                                              const float* __restrict__ pjw1, const float* __restrict__ pjb1,
                                              const float* __restrict__ pjw2, const float* __restrict__ pjb2,
                                              half_t* __restrict__ pf16) {
  int x = blockIdx.x;                 // 0..511
  int dom = x >> 8, i = x & 255;
  const float* src = emb + (size_t)((dom ? 2 : 0) * B_ + i) * 64;
  __shared__ float qrow[64];
  __shared__ float hbuf[HID_];
  __shared__ float obuf[64];
  __shared__ float nrm;
  int t = threadIdx.x;
  if (t < 64) qrow[t] = src[t];
  __syncthreads();
  float acc = pjb1[t];
  for (int c = 0; c < 64; ++c) acc += qrow[c] * pjw1[c * HID_ + t];
  hbuf[t] = fmaxf(acc, 0.f);
  __syncthreads();
  if (t < 64) {
    float o = pjb2[t];
    for (int h = 0; h < HID_; ++h) o += hbuf[h] * pjw2[h * 64 + t];
    obuf[t] = o;
  }
  __syncthreads();
  if (t == 0) {
    float ss = 0.f;
    for (int c = 0; c < 64; ++c) ss += obuf[c] * obuf[c];
    nrm = 1.f / fmaxf(sqrtf(ss), 1e-12f);
  }
  __syncthreads();
  if (t < 64) pf16[(size_t)(dom * B_ + i) * 64 + t] = (half_t)(obuf[t] * nrm);
}

// ---------------- kernel: WMMA GEMM  out[i][j] = scale * sum_c A[i][c]*Brow[j][c] ----------------
// B rows come from B0 (j < n0) else B1 (j - n0); boundary is tile aligned in all uses.
__global__ __launch_bounds__(128) void k_gemm(const half_t* __restrict__ A,
                                              const half_t* __restrict__ B0, int n0,
                                              const half_t* __restrict__ B1,
                                              float* __restrict__ out, int N, float scale) {
  int lane = threadIdx.x & 31, wave = threadIdx.x >> 5;
  int hh = lane >> 4, n = lane & 15;
  int jt = blockIdx.x * 4 + wave;
  int mt = blockIdx.y;
  int j = jt * 16 + n;
  int r = mt * 16 + n;
  const half_t* Arow = A + (size_t)r * 64;
  v16h a0 = load_afrag(Arow, 0, hh);
  v16h a1 = load_afrag(Arow, 1, hh);
  const half_t* Brow = (j < n0) ? (B0 + (size_t)j * 64) : (B1 + (size_t)(j - n0) * 64);
  v16h b0 = load_afrag(Brow, 0, hh);        // K 0..31 (same split order as A)
  v16h b1 = load_afrag(Brow, 1, hh);   // K 32..63
  v8f c = {};
  c = wmma_f16(a0, b0, c);
  c = wmma_f16(a1, b1, c);
#pragma unroll
  for (int v = 0; v < 8; ++v)
    out[(size_t)(mt * 16 + 8 * hh + v) * (size_t)N + j] = c[v] * scale;
}

// ---------------- kernel: nearest neighbor (argmax dot, unit-norm => argmin d2) ----------------
__global__ __launch_bounds__(256) void k_nn(const float* __restrict__ emb, float* __restrict__ lbl) {
  int i = blockIdx.x;
  const float* kt = emb + (size_t)(3 * B_ + i) * 64;  // k_t row
  __shared__ float krow[64];
  __shared__ float vbuf[256];
  __shared__ int ibuf[256];
  int t = threadIdx.x;
  if (t < 64) krow[t] = kt[t];
  __syncthreads();
  const float* qr = emb + (size_t)t * 64;             // q_s rows
  float dot = 0.f;
  for (int c = 0; c < 64; ++c) dot += krow[c] * qr[c];
  vbuf[t] = dot; ibuf[t] = t;
  __syncthreads();
  for (int s2 = 128; s2 > 0; s2 >>= 1) {
    if (t < s2) {
      float v2 = vbuf[t + s2]; int i2 = ibuf[t + s2];
      if (v2 > vbuf[t] || (v2 == vbuf[t] && i2 < ibuf[t])) { vbuf[t] = v2; ibuf[t] = i2; }
    }
    __syncthreads();
  }
  if (t == 0) lbl[i] = (float)ibuf[0];
}

// ---------------- kernel: domain discriminator ----------------
__global__ __launch_bounds__(256) void k_dom(const float* __restrict__ emb,
                                             const float* __restrict__ dw1, const float* __restrict__ db1,
                                             const float* __restrict__ dw2, const float* __restrict__ db2,
                                             float* __restrict__ out) {
  int row = blockIdx.x;  // 0..511 : [q_s ; q_t]
  const float* q = emb + (size_t)((row < B_) ? row : (2 * B_ + (row - B_))) * 64;
  __shared__ float qrow[64];
  __shared__ float red[256];
  int t = threadIdx.x;
  if (t < 64) qrow[t] = q[t];
  __syncthreads();
  float a = db1[t];
  for (int c = 0; c < 64; ++c) a += qrow[c] * dw1[c * HID_ + t];
  red[t] = fmaxf(a, 0.f) * dw2[t];
  __syncthreads();
  for (int s2 = 128; s2 > 0; s2 >>= 1) {
    if (t < s2) red[t] += red[t + s2];
    __syncthreads();
  }
  if (t == 0) out[row] = red[0] + db2[0];
}

// ---------------- kernel: regression head y_s ----------------
__global__ __launch_bounds__(256) void k_ys(const float* __restrict__ emb, const float* __restrict__ stat,
                                            const float* __restrict__ prw1, const float* __restrict__ prb1,
                                            const float* __restrict__ prw2, const float* __restrict__ prb2,
                                            float* __restrict__ out) {
  int i = blockIdx.x;
  __shared__ float xrow[96];
  __shared__ float r0[256], r1[256];
  int t = threadIdx.x;
  if (t < 64) xrow[t] = emb[(size_t)i * 64 + t];
  else if (t < 96) xrow[t] = stat[(size_t)i * 32 + (t - 64)];
  __syncthreads();
  float a = prb1[t];
  for (int c = 0; c < 96; ++c) a += xrow[c] * prw1[c * HID_ + t];
  a = fmaxf(a, 0.f);
  r0[t] = a * prw2[t * 2 + 0];
  r1[t] = a * prw2[t * 2 + 1];
  __syncthreads();
  for (int s2 = 128; s2 > 0; s2 >>= 1) {
    if (t < s2) { r0[t] += r0[t + s2]; r1[t] += r1[t + s2]; }
    __syncthreads();
  }
  if (t == 0) { out[i * 2 + 0] = r0[0] + prb2[0]; out[i * 2 + 1] = r1[0] + prb2[1]; }
}

// ---------------- kernel: integer / domain labels ----------------
__global__ __launch_bounds__(256) void k_labels(float* __restrict__ out) {
  int t = blockIdx.x * 256 + threadIdx.x;   // 0..1023
  if (t < 256)       out[O_LBL_S + t] = (float)t;
  else if (t < 512)  out[O_LBL_T + (t - 256)] = (float)(t - 256);
  else               out[O_LBL_DOM + (t - 512)] = ((t - 512) < 256) ? 1.f : 0.f;
}

// ---------------- workspace layout (bytes) ----------------
constexpr size_t WS_WF    = 0;                               // 240*512*2      = 245760
constexpr size_t WS_QTS   = WS_WF  + 240 * 512 * 2;          // 24576*64*2     = 3145728
constexpr size_t WS_QTT   = WS_QTS + (size_t)KQ_ * 64 * 2;
constexpr size_t WS_EMB32 = WS_QTT + (size_t)KQ_ * 64 * 2;   // 4*256*64*4
constexpr size_t WS_EMB16 = WS_EMB32 + (size_t)4 * B_ * 64 * 4;
constexpr size_t WS_PF16  = WS_EMB16 + (size_t)4 * B_ * 64 * 2;

extern "C" void kernel_launch(void* const* d_in, const int* in_sizes, int n_in,
                              void* d_out, int out_size, void* d_ws, size_t ws_size,
                              hipStream_t stream) {
  const float* sq_s  = (const float*)d_in[0];
  const float* sk_s  = (const float*)d_in[1];
  const float* stat_s= (const float*)d_in[2];
  const float* sq_t  = (const float*)d_in[3];
  const float* sk_t  = (const float*)d_in[4];
  const float* c1w   = (const float*)d_in[7];
  const float* c1b   = (const float*)d_in[8];
  const float* c2w   = (const float*)d_in[9];
  const float* c2b   = (const float*)d_in[10];
  const float* pjw1  = (const float*)d_in[11];
  const float* pjb1  = (const float*)d_in[12];
  const float* pjw2  = (const float*)d_in[13];
  const float* pjb2  = (const float*)d_in[14];
  const float* prw1  = (const float*)d_in[15];
  const float* prb1  = (const float*)d_in[16];
  const float* prw2  = (const float*)d_in[17];
  const float* prb2  = (const float*)d_in[18];
  const float* dw1   = (const float*)d_in[19];
  const float* db1   = (const float*)d_in[20];
  const float* dw2   = (const float*)d_in[21];
  const float* db2   = (const float*)d_in[22];
  const float* queue_s = (const float*)d_in[23];
  const float* queue_t = (const float*)d_in[24];

  float* out = (float*)d_out;
  char* ws = (char*)d_ws;
  half_t* wfrag  = (half_t*)(ws + WS_WF);
  half_t* qTs    = (half_t*)(ws + WS_QTS);
  half_t* qTt    = (half_t*)(ws + WS_QTT);
  float*  emb32  = (float*) (ws + WS_EMB32);
  half_t* emb16  = (half_t*)(ws + WS_EMB16);
  half_t* pf16   = (half_t*)(ws + WS_PF16);

  // packing
  k_pack_w<<<480, 256, 0, stream>>>(c1w, c2w, wfrag);
  k_pack_q<<<dim3(KQ_ / 32, 2), dim3(32, 8), 0, stream>>>(queue_s, qTs);
  k_pack_q<<<dim3(KQ_ / 32, 2), dim3(32, 8), 0, stream>>>(queue_t, qTt);

  // encoders (4 seq types x 256 batch)
  k_encode<<<4 * B_, 128, 0, stream>>>(sq_s, sk_s, sq_t, sk_t, wfrag, c1b, c2b, emb32, emb16);

  // projection MLPs -> p_q_s, p_q_t
  k_pmlp<<<2 * B_, 256, 0, stream>>>(emb32, pjw1, pjb1, pjw2, pjb2, pf16);

  // logits_s = [p_q_s @ k_s^T , p_q_s @ queue_s] / T
  k_gemm<<<dim3(NCOL / 64, B_ / 16), 128, 0, stream>>>(
      pf16, emb16 + (size_t)1 * B_ * 64, B_, qTs, out + O_LOGITS_S, NCOL, INV_T);
  // logits_t
  k_gemm<<<dim3(NCOL / 64, B_ / 16), 128, 0, stream>>>(
      pf16 + (size_t)B_ * 64, emb16 + (size_t)3 * B_ * 64, B_, qTt, out + O_LOGITS_T, NCOL, INV_T);
  // logits_ts = q_t @ q_s^T / T
  k_gemm<<<dim3(B_ / 64, B_ / 16), 128, 0, stream>>>(
      emb16 + (size_t)2 * B_ * 64, emb16, B_, emb16, out + O_LOGITS_TS, B_, INV_T);

  // nearest-neighbor labels, domain head, regression head, labels
  k_nn<<<B_, 256, 0, stream>>>(emb32, out + O_LBL_TS);
  k_dom<<<2 * B_, 256, 0, stream>>>(emb32, dw1, db1, dw2, db2, out + O_PRED_DOM);
  k_ys<<<B_, 256, 0, stream>>>(emb32, stat_s, prw1, prb1, prw2, prb2, out + O_YS);
  k_labels<<<4, 256, 0, stream>>>(out);
}
